// FC2_P_3719441679193
// MI455X (gfx1250) — matrix operand order repaired and checked by reading, not compiled, not run on hardware
//
#include <hip/hip_runtime.h>
#include <hip/hip_bf16.h>
#include <stdint.h>

// ---------------- problem constants (from reference) ----------------
#define CLEN   111            // CONTEXT_LENGTH
#define ILEN   51             // INST_LENGTH
#define PCN    256            // PC
#define F1N    400            // F1
#define BATCH  8192
#define TSTEPS 110            // CLEN - 1
#define KPAD   64             // ILEN padded to multiple of 32 for 16x16x32 wmma
#define K2     (PCN * TSTEPS) // 28160 : K of FC1 (permuted order k' = t*256 + o)
#define NT2    (F1N / 16)     // 25 N-tiles of FC1
#define K2STEPS (K2 / 32)     // 880 K-steps of FC1

typedef __attribute__((ext_vector_type(16))) __bf16 v16bf;
typedef __attribute__((ext_vector_type(8)))  __bf16 v8bf;
typedef __attribute__((ext_vector_type(8)))  float  v8f;

static __device__ __forceinline__ v16bf cat8(v8bf lo, v8bf hi) {
  return __builtin_shufflevector(lo, hi, 0,1,2,3,4,5,6,7,8,9,10,11,12,13,14,15);
}

// ---------------- prep kernels: fp32 -> bf16 with pad/permute ----------------

// xb[b, t, kk] = bf16(x[b, t, kk]) for kk<51, else 0.   (x flat index == bt*51+kk)
__global__ __launch_bounds__(256) void prep_x_kernel(const float* __restrict__ x,
                                                     __bf16* __restrict__ xb) {
  const size_t total = (size_t)BATCH * CLEN * KPAD;
  const size_t step  = (size_t)gridDim.x * blockDim.x;
  for (size_t i = (size_t)blockIdx.x * blockDim.x + threadIdx.x; i < total; i += step) {
    const int    kk = (int)(i & (KPAD - 1));
    const size_t bt = i >> 6;                 // b*CLEN + t
    float v = 0.0f;
    if (kk < ILEN) v = x[bt * ILEN + kk];
    xb[i] = (__bf16)v;
  }
}

// wc0/wc1[o, kk] = bf16(Wc[o, kk, 0/1]) padded to KPAD
__global__ __launch_bounds__(256) void prep_wc_kernel(const float* __restrict__ Wc,
                                                      __bf16* __restrict__ wc0,
                                                      __bf16* __restrict__ wc1) {
  const int i = blockIdx.x * blockDim.x + threadIdx.x;
  if (i >= PCN * KPAD) return;
  const int o = i >> 6, kk = i & (KPAD - 1);
  float v0 = 0.0f, v1 = 0.0f;
  if (kk < ILEN) {
    v0 = Wc[(o * ILEN + kk) * 2 + 0];
    v1 = Wc[(o * ILEN + kk) * 2 + 1];
  }
  wc0[i] = (__bf16)v0;
  wc1[i] = (__bf16)v1;
}

// w1p[n, t*256 + o] = bf16(W1[n, o*110 + t])   (K permutation shared with y)
__global__ __launch_bounds__(256) void prep_w1_kernel(const float* __restrict__ W1,
                                                      __bf16* __restrict__ w1p) {
  const size_t total = (size_t)F1N * K2;
  const size_t step  = (size_t)gridDim.x * blockDim.x;
  for (size_t i = (size_t)blockIdx.x * blockDim.x + threadIdx.x; i < total; i += step) {
    const int n  = (int)(i / K2);
    const int r  = (int)(i % K2);
    const int tt = r >> 8;           // / 256
    const int o  = r & 255;
    w1p[i] = (__bf16)W1[(size_t)n * K2 + (size_t)o * TSTEPS + tt];
  }
}

// ---------------- stage 1: conv-as-GEMM, fused bias+ReLU, y in bf16 ----------------
// Block: 256 thr (8 waves), covers 16 batch rows. Wave w handles o-tiles {w, w+8}.
// y[b, tt*256 + o] = relu(x0·Wc0 + bc + x_{tt+1}·Wc1)
__global__ __launch_bounds__(256) void conv_stage_kernel(const __bf16* __restrict__ xb,
                                                         const __bf16* __restrict__ wc0,
                                                         const __bf16* __restrict__ wc1,
                                                         const float*  __restrict__ bc,
                                                         __bf16* __restrict__ Y) {
  const int lane  = threadIdx.x & 31;
  const int wv    = threadIdx.x >> 5;
  const int lrow  = lane & 15;
  const int hi    = lane >> 4;
  const int bbase = blockIdx.x * 16;

  const int ot[2] = { wv, wv + 8 };

  // B fragments (Wc columns) — constant over the t loop.
  v16bf bf0[2][2], bf1[2][2];
#pragma unroll
  for (int i = 0; i < 2; ++i) {
    const __bf16* w0r = wc0 + (size_t)(ot[i] * 16 + lrow) * KPAD + hi * 16;
    const __bf16* w1r = wc1 + (size_t)(ot[i] * 16 + lrow) * KPAD + hi * 16;
#pragma unroll
    for (int ks = 0; ks < 2; ++ks) {
      bf0[i][ks] = *(const v16bf*)(w0r + ks * 32);
      bf1[i][ks] = *(const v16bf*)(w1r + ks * 32);
    }
  }

  // A row base for this lane (row = batch b, within xb[b, t, 64])
  const __bf16* arow = xb + (size_t)(bbase + lrow) * (CLEN * KPAD) + hi * 16;

  // z0 = x_{t=0} · Wc0 + bc  (seed accumulator, reused for every t)
  v16bf a0[2];
#pragma unroll
  for (int ks = 0; ks < 2; ++ks)
    a0[ks] = cat8(*(const v8bf*)(arow + ks * 32), *(const v8bf*)(arow + ks * 32 + 8));

  const v8f vzero = {};
  v8f z0[2];
#pragma unroll
  for (int i = 0; i < 2; ++i) {
    v8f acc = vzero;
    acc = __builtin_amdgcn_wmma_f32_16x16x32_bf16(false, a0[0], false, bf0[i][0], (short)0, acc, false, false);
    acc = __builtin_amdgcn_wmma_f32_16x16x32_bf16(false, a0[1], false, bf0[i][1], (short)0, acc, false, false);
    const float bcv = bc[ot[i] * 16 + lrow];
#pragma unroll
    for (int r = 0; r < 8; ++r) acc[r] += bcv;
    z0[i] = acc;
  }

  for (int tt = 0; tt < TSTEPS; ++tt) {
    const __bf16* ar = arow + (size_t)(tt + 1) * KPAD;
    v16bf a[2];
#pragma unroll
    for (int ks = 0; ks < 2; ++ks)
      a[ks] = cat8(*(const v8bf*)(ar + ks * 32), *(const v8bf*)(ar + ks * 32 + 8));

#pragma unroll
    for (int i = 0; i < 2; ++i) {
      v8f acc = z0[i];
      acc = __builtin_amdgcn_wmma_f32_16x16x32_bf16(false, a[0], false, bf1[i][0], (short)0, acc, false, false);
      acc = __builtin_amdgcn_wmma_f32_16x16x32_bf16(false, a[1], false, bf1[i][1], (short)0, acc, false, false);
      // relu + store bf16 at y[b = bbase+hi*8+r, k' = tt*256 + ot*16 + lrow]
      __bf16* yp = Y + (size_t)(bbase + hi * 8) * K2 + (size_t)tt * PCN + ot[i] * 16 + lrow;
#pragma unroll
      for (int r = 0; r < 8; ++r) {
        float v = acc[r];
        v = v > 0.0f ? v : 0.0f;
        yp[(size_t)r * K2] = (__bf16)v;
      }
    }
  }
}

// ---------------- stage 2: FC1 (the big GEMM), fused bias+ReLU ----------------
// Block: 128 thr (4 waves). Wave owns one 16-row M tile x all 25 N tiles.
// y streamed from HBM once; W1p (22.5 MB bf16) served from L2.
__global__ __launch_bounds__(128) void fc1_stage_kernel(const __bf16* __restrict__ Y,
                                                        const __bf16* __restrict__ W1p,
                                                        const float*  __restrict__ b1,
                                                        float* __restrict__ H) {
  const int lane  = threadIdx.x & 31;
  const int wv    = threadIdx.x >> 5;
  const int lrow  = lane & 15;
  const int hi    = lane >> 4;
  const int mbase = (blockIdx.x * 4 + wv) * 16;

  const v8f vzero = {};
  v8f acc[NT2];
#pragma unroll
  for (int nt = 0; nt < NT2; ++nt) acc[nt] = vzero;

  const __bf16* aptr  = Y   + (size_t)(mbase + lrow) * K2 + hi * 16;
  const __bf16* bptr0 = W1p + (size_t)lrow * K2 + hi * 16;

  for (int kb = 0; kb < K2STEPS; ++kb) {
    const __bf16* ap = aptr + kb * 32;
    __builtin_prefetch(ap + 256, 0, 0);           // stream-ahead on the A (y) stream
    const v16bf a = cat8(*(const v8bf*)(ap), *(const v8bf*)(ap + 8));
    const __bf16* bp = bptr0 + kb * 32;
#pragma unroll
    for (int nt = 0; nt < NT2; ++nt) {
      const v16bf b = *(const v16bf*)(bp + (size_t)(nt * 16) * K2);
      acc[nt] = __builtin_amdgcn_wmma_f32_16x16x32_bf16(false, a, false, b, (short)0, acc[nt], false, false);
    }
  }

#pragma unroll
  for (int nt = 0; nt < NT2; ++nt) {
    const int n = nt * 16 + lrow;
    const float bias = b1[n];
    float* hp = H + (size_t)(mbase + hi * 8) * F1N + n;
#pragma unroll
    for (int r = 0; r < 8; ++r) {
      float v = acc[nt][r] + bias;
      hp[(size_t)r * F1N] = v > 0.0f ? v : 0.0f;
    }
  }
}

// ---------------- stage 3: FC2 (tiny) — one wave per batch row ----------------
__global__ __launch_bounds__(256) void fc2_stage_kernel(const float* __restrict__ H,
                                                        const float* __restrict__ W2,
                                                        const float* __restrict__ b2,
                                                        float* __restrict__ out) {
  const int lane = threadIdx.x & 31;
  const int wv   = threadIdx.x >> 5;
  const int b    = blockIdx.x * 8 + wv;
  const float* hp = H + (size_t)b * F1N;

  float s0 = 0.0f, s1 = 0.0f;
  for (int n = lane; n < F1N; n += 32) {
    const float hv = hp[n];
    s0 += hv * W2[n];
    s1 += hv * W2[F1N + n];
  }
#pragma unroll
  for (int off = 16; off > 0; off >>= 1) {
    s0 += __shfl_xor(s0, off, 32);
    s1 += __shfl_xor(s1, off, 32);
  }
  if (lane == 0) {
    out[b * 2 + 0] = s0 + b2[0];
    out[b * 2 + 1] = s1 + b2[1];
  }
}

// ---------------- launcher ----------------
extern "C" void kernel_launch(void* const* d_in, const int* in_sizes, int n_in,
                              void* d_out, int out_size, void* d_ws, size_t ws_size,
                              hipStream_t stream) {
  const float* x  = (const float*)d_in[0];
  const float* Wc = (const float*)d_in[1];
  const float* bc = (const float*)d_in[2];
  const float* W1 = (const float*)d_in[3];
  const float* b1 = (const float*)d_in[4];
  const float* W2 = (const float*)d_in[5];
  const float* b2 = (const float*)d_in[6];
  float* out = (float*)d_out;

  // workspace carve-up (all offsets 256B-aligned)
  char* ws = (char*)d_ws;
  const size_t XB_BYTES  = (size_t)BATCH * CLEN * KPAD * 2;  // 116,391,936
  const size_t WC_BYTES  = (size_t)PCN * KPAD * 2;           // 32,768
  const size_t W1P_BYTES = (size_t)F1N * K2 * 2;             // 22,528,000
  const size_t Y_BYTES   = (size_t)BATCH * K2 * 2;           // 461,373,440

  __bf16* xb  = (__bf16*)(ws);
  __bf16* wc0 = (__bf16*)(ws + XB_BYTES);
  __bf16* wc1 = (__bf16*)(ws + XB_BYTES + WC_BYTES);
  __bf16* w1p = (__bf16*)(ws + XB_BYTES + 2 * WC_BYTES);
  __bf16* Yb  = (__bf16*)(ws + XB_BYTES + 2 * WC_BYTES + W1P_BYTES);
  float*  H   = (float*)(ws + XB_BYTES + 2 * WC_BYTES + W1P_BYTES + Y_BYTES);

  prep_x_kernel <<<4096, 256, 0, stream>>>(x, xb);
  prep_wc_kernel<<<(PCN * KPAD + 255) / 256, 256, 0, stream>>>(Wc, wc0, wc1);
  prep_w1_kernel<<<4096, 256, 0, stream>>>(W1, w1p);

  conv_stage_kernel<<<BATCH / 16, 256, 0, stream>>>(xb, wc0, wc1, bc, Yb);
  fc1_stage_kernel <<<BATCH / 64, 128, 0, stream>>>(Yb, w1p, b1, H);
  fc2_stage_kernel <<<BATCH / 8, 256, 0, stream>>>(H, W2, b2, out);
}
